// CompressedMoEBlock_45535243272669
// MI455X (gfx1250) — matrix operand order repaired and checked by reading, not compiled, not run on hardware
//
#include <hip/hip_runtime.h>
#include <cstdint>
#include <cstddef>

// ---------------------------------------------------------------------------
// Types for CDNA5 WMMA (wave32, gfx1250)
// ---------------------------------------------------------------------------
typedef __bf16 bf16_t;
typedef bf16_t v16bf __attribute__((ext_vector_type(16)));
typedef float  v8f   __attribute__((ext_vector_type(8)));

#define TM 128
#define TN 128
#define TK 32

__device__ __forceinline__ unsigned short f2bf(float f) {
    // round-to-nearest-even fp32 -> bf16 (bit-level)
    unsigned int u = __float_as_uint(f);
    unsigned int r = u + 0x7FFFu + ((u >> 16) & 1u);
    return (unsigned short)(r >> 16);
}

// Generic pointer (to a __shared__ object) -> 32-bit LDS byte offset.
__device__ __forceinline__ unsigned lds_off(const void* p) {
    return (unsigned)(unsigned long long)
        (__attribute__((address_space(3))) const void*)p;
}

// ---------------------------------------------------------------------------
// Generic tiled bf16 GEMM:  out[M,N] (+)= A[M,K] * B[K,N]
// A row-major bf16 [M,K]; Bt = B transposed, row-major bf16 [N,K].
// Tiles staged with GLOBAL_LOAD_ASYNC_TO_LDS_B128 (ASYNCcnt), double-buffered:
// one s_wait_asynccnt + one barrier per K-step. All 8 B fragments are
// prefetched into registers before the back-to-back WMMA chain.
// mode 0: store bf16            mode 1: store f32
// mode 2: out_bf16 = silu(aux_f32[idx]) * acc
// mode 3: out_f32[idx] += combine[row*8+expert] * acc
// ---------------------------------------------------------------------------
__global__ __launch_bounds__(256)
void gemm_bf16_kernel(const unsigned short* __restrict__ A,
                      const unsigned short* __restrict__ Bt,
                      void* __restrict__ outp,
                      const float* __restrict__ aux,
                      const float* __restrict__ combine,
                      int expert, int M, int N, int K, int mode)
{
    __shared__ alignas(16) unsigned short lA[2 * TM * TK];
    __shared__ alignas(16) unsigned short lB[2 * TN * TK];

    const int tid  = threadIdx.x;
    const int wave = tid >> 5;        // 0..7, each wave owns a 16-row strip
    const int lane = tid & 31;
    const int l16  = lane & 15;
    const int hi   = lane >> 4;       // half-wave select
    const int m0   = blockIdx.y * TM;
    const int n0   = blockIdx.x * TN;

    const unsigned baseA = lds_off(lA);
    const unsigned baseB = lds_off(lB);

    // Issue one K-tile (128x32 bf16 for A and Bt) as 4 async b128 loads/thread.
    // A's and Bt's 8-element K-chunks are permuted [0,2,1,3] so each lane's WMMA
    // fragment is one contiguous 32B LDS read.
    auto issue_tile = [&](int kt, int buf) {
#pragma unroll
        for (int i = 0; i < 2; ++i) {
            const int q = tid + 256 * i;
            const int r = q >> 2;      // tile row (0..127)
            const int c = q & 3;       // 8-element K chunk
            const int lofs = ((c & 1) << 4) | ((c >> 1) << 3);
            const unsigned la = baseA + (unsigned)((buf * TM * TK + r * TK + lofs) * 2);
            const void* ga = (const void*)(A + (size_t)(m0 + r) * K + kt + 8 * c);
            asm volatile("global_load_async_to_lds_b128 %0, %1, off"
                         :: "v"(la), "v"(ga) : "memory");
            const unsigned lb = baseB + (unsigned)((buf * TN * TK + r * TK + lofs) * 2);
            const void* gb = (const void*)(Bt + (size_t)(n0 + r) * K + kt + 8 * c);
            asm volatile("global_load_async_to_lds_b128 %0, %1, off"
                         :: "v"(lb), "v"(gb) : "memory");
        }
    };

    v8f acc[8];
#pragma unroll
    for (int j = 0; j < 8; ++j)
#pragma unroll
        for (int i = 0; i < 8; ++i) acc[j][i] = 0.0f;

    issue_tile(0, 0);
    for (int kt = 0; kt < K; kt += TK) {
        const int buf = (kt / TK) & 1;
        // Publish this tile (own async ops done + block-wide barrier). The
        // barrier also guarantees everyone finished reading the buffer we are
        // about to overwrite with the next issue.
        asm volatile("s_wait_asynccnt 0x0" ::: "memory");
        __syncthreads();
        if (kt + TK < K) issue_tile(kt + TK, buf ^ 1);

        const unsigned short* pA = lA + buf * TM * TK;
        const unsigned short* pB = lB + buf * TN * TK;

        // Prefetch all fragments, then run the WMMA chain back-to-back
        // (independent accumulators -> no D->A/B hazards, full co-execution).
        const v16bf a = *(const v16bf*)(pA + (wave * 16 + l16) * TK + hi * 16);
        v16bf b[8];
#pragma unroll
        for (int j = 0; j < 8; ++j)
            b[j] = *(const v16bf*)(pB + (j * 16 + l16) * TK + hi * 16);
#pragma unroll
        for (int j = 0; j < 8; ++j)
            acc[j] = __builtin_amdgcn_wmma_f32_16x16x32_bf16(
                false, a, false, b[j], (short)0, acc[j], false, false);
    }

    // Epilogue. C/D layout: VGPR r, lanes 0-15 -> M=r, lanes 16-31 -> M=r+8.
#pragma unroll
    for (int j = 0; j < 8; ++j) {
        const int col = n0 + j * 16 + l16;
#pragma unroll
        for (int r = 0; r < 8; ++r) {
            const int row   = m0 + wave * 16 + hi * 8 + r;
            const float v   = acc[j][r];
            const size_t ix = (size_t)row * N + col;
            if (mode == 0) {
                ((unsigned short*)outp)[ix] = f2bf(v);
            } else if (mode == 1) {
                ((float*)outp)[ix] = v;
            } else if (mode == 2) {
                const float g = aux[ix];
                const float s = g / (1.0f + __expf(-g));
                ((unsigned short*)outp)[ix] = f2bf(s * v);
            } else {
                const float c = combine[(size_t)row * 8 + expert];
                float* o = (float*)outp;
                o[ix] += c * v;
            }
        }
    }
}

// ---------------------------------------------------------------------------
// Gating: logits = x @ gate_w, softmax, top-2, renormalize -> combine[T,8]
// One wave (32 lanes) per token.
// ---------------------------------------------------------------------------
__global__ __launch_bounds__(256)
void gate_kernel(const float* __restrict__ x, const float* __restrict__ gw,
                 float* __restrict__ combine, int T, int H)
{
    const int wave = threadIdx.x >> 5;
    const int lane = threadIdx.x & 31;
    const int t = blockIdx.x * 8 + wave;
    if (t >= T) return;

    float acc[8];
#pragma unroll
    for (int e = 0; e < 8; ++e) acc[e] = 0.0f;
    for (int h = lane; h < H; h += 32) {
        const float xv = x[(size_t)t * H + h];
#pragma unroll
        for (int e = 0; e < 8; ++e) acc[e] += xv * gw[h * 8 + e];
    }
#pragma unroll
    for (int e = 0; e < 8; ++e)
        for (int m = 16; m >= 1; m >>= 1)
            acc[e] += __shfl_xor(acc[e], m, 32);

    float mx = acc[0];
#pragma unroll
    for (int e = 1; e < 8; ++e) mx = fmaxf(mx, acc[e]);
    float p[8];
#pragma unroll
    for (int e = 0; e < 8; ++e) p[e] = __expf(acc[e] - mx);

    int a1 = 0;
#pragma unroll
    for (int e = 1; e < 8; ++e) if (p[e] > p[a1]) a1 = e;
    int a2 = (a1 == 0) ? 1 : 0;
#pragma unroll
    for (int e = 0; e < 8; ++e) if (e != a1 && p[e] > p[a2]) a2 = e;

    const float denom = p[a1] + p[a2];
    if (lane < 8) {
        const float c = (lane == a1) ? p[a1] / denom
                      : (lane == a2) ? p[a2] / denom : 0.0f;
        combine[(size_t)t * 8 + lane] = c;
    }
}

// ---------------------------------------------------------------------------
// Converters & init
// ---------------------------------------------------------------------------
__global__ void cvt_kernel(const float* __restrict__ src,
                           unsigned short* __restrict__ dst, int n)
{
    const int i = blockIdx.x * 256 + threadIdx.x;
    if (i < n) dst[i] = f2bf(src[i]);
}

// src [K,N] fp32 -> dst [N,K] bf16, 32x32 LDS-tiled (coalesced both sides).
__global__ __launch_bounds__(256)
void cvt_t_kernel(const float* __restrict__ src,
                  unsigned short* __restrict__ dst, int K, int N)
{
    __shared__ unsigned short tile[32][33];
    const int k0 = blockIdx.y * 32, n0 = blockIdx.x * 32;
    const int ln = threadIdx.x & 31, w = threadIdx.x >> 5;  // 8 waves
#pragma unroll
    for (int i = 0; i < 4; ++i) {
        const int k = w * 4 + i;
        tile[k][ln] = f2bf(src[(size_t)(k0 + k) * N + n0 + ln]);
    }
    __syncthreads();
#pragma unroll
    for (int i = 0; i < 4; ++i) {
        const int n = w * 4 + i;
        dst[(size_t)(n0 + n) * K + k0 + ln] = tile[ln][n];
    }
}

__global__ void zero_kernel(float* __restrict__ p, int n)
{
    const int i = blockIdx.x * 256 + threadIdx.x;
    if (i < n) p[i] = 0.0f;
}

// ---------------------------------------------------------------------------
// Host-side orchestration
// ---------------------------------------------------------------------------
extern "C" void kernel_launch(void* const* d_in, const int* in_sizes, int n_in,
                              void* d_out, int out_size, void* d_ws, size_t ws_size,
                              hipStream_t stream)
{
    (void)in_sizes; (void)n_in; (void)ws_size;
    const int E = 8, H = 1024, F = 2816, R = 256, T = 2048;

    const float* x      = (const float*)d_in[0];
    const float* gate_w = (const float*)d_in[1];
    const float* Ug     = (const float*)d_in[2];
    const float* Cg     = (const float*)d_in[3];
    const float* Vg     = (const float*)d_in[4];
    const float* Uu     = (const float*)d_in[5];
    const float* Cu     = (const float*)d_in[6];
    const float* Vu     = (const float*)d_in[7];
    const float* Ud     = (const float*)d_in[8];
    const float* Cd     = (const float*)d_in[9];
    const float* Vd     = (const float*)d_in[10];
    float* out = (float*)d_out;

    // workspace carve-up (256B aligned)
    char* ws = (char*)d_ws;
    size_t off = 0;
    auto carve = [&](size_t bytes) -> char* {
        off = (off + 255) & ~(size_t)255;
        char* p = ws + off;
        off += bytes;
        return p;
    };
    unsigned short* x16     = (unsigned short*)carve((size_t)T * H * 2);
    float*          combine = (float*)         carve((size_t)T * E * 4);
    unsigned short* Cgt     = (unsigned short*)carve((size_t)R * R * 2);
    unsigned short* Cut     = (unsigned short*)carve((size_t)R * R * 2);
    unsigned short* Cdt     = (unsigned short*)carve((size_t)R * R * 2);
    unsigned short* wbuf    = (unsigned short*)carve((size_t)F * R * 2); // staged weight
    unsigned short* Tg      = (unsigned short*)carve((size_t)T * R * 2);
    unsigned short* Tg2     = (unsigned short*)carve((size_t)T * R * 2);
    unsigned short* Tu      = (unsigned short*)carve((size_t)T * R * 2);
    unsigned short* Tu2     = (unsigned short*)carve((size_t)T * R * 2);
    unsigned short* Q       = (unsigned short*)carve((size_t)T * R * 2);
    unsigned short* Q2      = (unsigned short*)carve((size_t)T * R * 2);
    float*          Gbuf    = (float*)         carve((size_t)T * F * 4);
    unsigned short* Aact    = (unsigned short*)carve((size_t)T * F * 2);

    auto cvt = [&](const float* s, unsigned short* d, int n) {
        cvt_kernel<<<(n + 255) / 256, 256, 0, stream>>>(s, d, n);
    };
    auto cvt_t = [&](const float* s, unsigned short* d, int Kd, int Nd) {
        dim3 g(Nd / 32, Kd / 32);
        cvt_t_kernel<<<g, 256, 0, stream>>>(s, d, Kd, Nd);
    };
    auto gemm = [&](const unsigned short* A, const unsigned short* Bt, void* o,
                    const float* aux, const float* comb, int e,
                    int M, int N, int Kd, int mode) {
        dim3 grid(N / TN, M / TM);
        gemm_bf16_kernel<<<grid, 256, 0, stream>>>(A, Bt, o, aux, comb, e,
                                                   M, N, Kd, mode);
    };

    // prologue
    gate_kernel<<<T / 8, 256, 0, stream>>>(x, gate_w, combine, T, H);
    cvt(x, x16, T * H);
    cvt_t(Cg, Cgt, R, R);
    cvt_t(Cu, Cut, R, R);
    cvt_t(Cd, Cdt, R, R);
    zero_kernel<<<(out_size + 255) / 256, 256, 0, stream>>>(out, out_size);

    for (int e = 0; e < E; ++e) {
        const float* Uge = Ug + (size_t)e * H * R;   // [H,R]
        const float* Vge = Vg + (size_t)e * R * F;   // [R,F]
        const float* Uue = Uu + (size_t)e * H * R;
        const float* Vue = Vu + (size_t)e * R * F;
        const float* Ude = Ud + (size_t)e * F * R;   // [F,R]
        const float* Vde = Vd + (size_t)e * R * H;   // [R,H]

        // gate path: Tg2 = (x @ Ug[e]) @ Cg
        cvt_t(Uge, wbuf, H, R);
        gemm(x16, wbuf, Tg,  nullptr, nullptr, e, T, R, H, 0);
        gemm(Tg,  Cgt,  Tg2, nullptr, nullptr, e, T, R, R, 0);

        // up path: Tu2 = (x @ Uu[e]) @ Cu
        cvt_t(Uue, wbuf, H, R);
        gemm(x16, wbuf, Tu,  nullptr, nullptr, e, T, R, H, 0);
        gemm(Tu,  Cut,  Tu2, nullptr, nullptr, e, T, R, R, 0);

        // g = Tg2 @ Vg[e]  (fp32),  a = silu(g) * (Tu2 @ Vu[e])  (bf16)
        cvt_t(Vge, wbuf, R, F);
        gemm(Tg2, wbuf, Gbuf, nullptr, nullptr, e, T, F, R, 1);
        cvt_t(Vue, wbuf, R, F);
        gemm(Tu2, wbuf, Aact, Gbuf,   nullptr, e, T, F, R, 2);

        // down path: out += combine[:,e] * ((a @ Ud[e]) @ Cd @ Vd[e])
        cvt_t(Ude, wbuf, F, R);
        gemm(Aact, wbuf, Q,  nullptr, nullptr, e, T, R, F, 0);
        gemm(Q,    Cdt,  Q2, nullptr, nullptr, e, T, R, R, 0);
        cvt_t(Vde, wbuf, R, H);
        gemm(Q2,   wbuf, out, nullptr, combine, e, T, H, R, 3);
    }
}
